// SurrogateAttentionBlock_72662256714523
// MI455X (gfx1250) — matrix operand order repaired and checked by reading, not compiled, not run on hardware
//
#include <hip/hip_runtime.h>

// ---------------------------------------------------------------- types
typedef unsigned short u16_t;
typedef __attribute__((ext_vector_type(16))) __bf16 v16bf;
typedef __attribute__((ext_vector_type(8)))  __bf16 v8bf;
typedef __attribute__((ext_vector_type(8)))  float   v8f;
typedef __attribute__((ext_vector_type(4)))  unsigned int v4u;
typedef __attribute__((ext_vector_type(8)))  int v8i;
typedef __attribute__((ext_vector_type(4)))  int v4i;

#define S_LEN   8192
#define D_DIM   512
#define NHEAD   8
#define HDIM    64
#define NSUR    32
#define NBATCH  8
#define NROWS   (NBATCH * S_LEN)         /* 65536 */
#define SCALE_INV 0.125f                 /* 1/sqrt(64) */
#define NPART   16
#define KEYS_PER_PART (S_LEN / NPART)    /* 512 */
#define CHUNKS  (KEYS_PER_PART / 32)     /* 16  */
#define LN_EPS  1e-5f

// ---------------------------------------------------------------- helpers
__device__ __forceinline__ unsigned short f2bf(float f) {
  unsigned u = __float_as_uint(f);
  unsigned r = (u + 0x7FFFu + ((u >> 16) & 1u)) >> 16;
  return (unsigned short)r;
}

__device__ __forceinline__ v8f zero_v8f() {
  v8f z = {0.f, 0.f, 0.f, 0.f, 0.f, 0.f, 0.f, 0.f};
  return z;
}

__device__ __forceinline__ v8f wmma_bf16(v16bf a, v16bf b, v8f c) {
  return __builtin_amdgcn_wmma_f32_16x16x32_bf16(false, a, false, b, (short)0, c,
                                                 false, false);
}

// A-fragment, 16x32 bf16 tile at (m0,k0) of row-major [*, ld] matrix.
// Lane L (m=L&15, half=L>>4) holds K = half*16 + 0..15 (contiguous).
__device__ __forceinline__ v16bf load_a_frag(const u16_t* base, int ld, int m0, int k0) {
  int lane = threadIdx.x & 31;
  const u16_t* p = base + (size_t)(m0 + (lane & 15)) * ld + k0 + ((lane >> 4) << 4);
  v8bf lo = *reinterpret_cast<const v8bf*>(p);
  v8bf hi = *reinterpret_cast<const v8bf*>(p + 8);
  return __builtin_shufflevector(lo, hi, 0, 1, 2, 3, 4, 5, 6, 7, 8, 9, 10, 11, 12, 13, 14, 15);
}

// B-fragment: B is (K x N) = Wt; we load from W stored as [N, K] row-major
// (exactly nn.Linear weight layout).  Lane L: column n = n0 + (L&15),
// K = k0 + (L>>4)*16 + 0..15 (contiguous).
__device__ __forceinline__ v16bf load_b_frag(const u16_t* w, int ld, int n0, int k0) {
  int lane = threadIdx.x & 31;
  const u16_t* p = w + (size_t)(n0 + (lane & 15)) * ld + k0 + ((lane >> 4) << 4);
  v8bf lo = *reinterpret_cast<const v8bf*>(p);
  v8bf hi = *reinterpret_cast<const v8bf*>(p + 8);
  return __builtin_shufflevector(lo, hi, 0, 1, 2, 3, 4, 5, 6, 7, 8, 9, 10, 11, 12, 13, 14, 15);
}

// ---------------------------------------------------------------- K0: f32 -> bf16
__global__ void cvt_bf16_kernel(const float* __restrict__ src, u16_t* __restrict__ dst, int n) {
  int i = (blockIdx.x * blockDim.x + threadIdx.x) * 4;
  if (i + 3 < n) {
    float4 f = *reinterpret_cast<const float4*>(src + i);
    dst[i + 0] = f2bf(f.x);
    dst[i + 1] = f2bf(f.y);
    dst[i + 2] = f2bf(f.z);
    dst[i + 3] = f2bf(f.w);
  } else {
    for (int j = i; j < n; ++j) dst[j] = f2bf(src[j]);
  }
}

// ---------------------------------------------------------------- K1: C = A @ W^T + bias (bf16 in/out, f32 accum)
// block: 256 threads = 8 waves (2 row-groups x 4 col-groups); block tile 32x128.
__global__ void __launch_bounds__(256) gemm_bias_kernel(
    const u16_t* __restrict__ A, const u16_t* __restrict__ W,
    const float* __restrict__ bias, u16_t* __restrict__ C, int M) {
  const int wv = threadIdx.x >> 5;
  const int m0 = blockIdx.x * 32 + (wv >> 2) * 16;
  const int n0 = blockIdx.y * 128 + (wv & 3) * 32;
  if (m0 >= M) return;  // uniform per wave; kernel has no barriers

  v8f c0 = zero_v8f(), c1 = zero_v8f();
#pragma unroll
  for (int k0 = 0; k0 < D_DIM; k0 += 32) {
    v16bf a  = load_a_frag(A, D_DIM, m0, k0);
    v16bf b0 = load_b_frag(W, D_DIM, n0, k0);
    v16bf b1 = load_b_frag(W, D_DIM, n0 + 16, k0);
    c0 = wmma_bf16(a, b0, c0);
    c1 = wmma_bf16(a, b1, c1);
  }
  const int lane = threadIdx.x & 31, half = lane >> 4, col = lane & 15;
#pragma unroll
  for (int v = 0; v < 8; ++v) {
    size_t row = (size_t)(m0 + v + half * 8) * D_DIM;
    C[row + n0 + col]      = f2bf(c0[v] + bias[n0 + col]);
    C[row + n0 + 16 + col] = f2bf(c1[v] + bias[n0 + 16 + col]);
  }
}

// ---------------------------------------------------------------- K2: stage-1 compress attention (flash, partitioned)
// grid: (B*H, NPART), block: 32 (one wave). Q: qc[32, 512] (batch-invariant).
__global__ void __launch_bounds__(32) compress_attn_kernel(
    const u16_t* __restrict__ qc, const u16_t* __restrict__ kc,
    const u16_t* __restrict__ vc, float* __restrict__ part_o,
    float* __restrict__ part_ml) {
  __shared__ u16_t klds[32 * 64];
  __shared__ u16_t vlds[32 * 72];     // padded rows for the transpose gather
  __shared__ float sclds[32 * 33];    // padded scores
  __shared__ u16_t plds[32 * 48];     // padded P (16B-aligned rows)
  __shared__ float scl_lds[32];

  const int bh = blockIdx.x, part = blockIdx.y;
  const int b = bh >> 3, h = bh & 7;
  const int lane = threadIdx.x;
  const int half = lane >> 4, col = lane & 15;

  v16bf aq[2][2];
#pragma unroll
  for (int mi = 0; mi < 2; ++mi)
#pragma unroll
    for (int kg = 0; kg < 2; ++kg)
      aq[mi][kg] = load_a_frag(qc, D_DIM, mi * 16, h * HDIM + kg * 32);

  v8f o[2][4];
#pragma unroll
  for (int mi = 0; mi < 2; ++mi)
#pragma unroll
    for (int nj = 0; nj < 4; ++nj) o[mi][nj] = zero_v8f();
  float m_r = -1e30f, l_r = 0.f;

  const u16_t* kbase = kc + (size_t)b * S_LEN * D_DIM + h * HDIM;
  const u16_t* vbase = vc + (size_t)b * S_LEN * D_DIM + h * HDIM;

  for (int c = 0; c < CHUNKS; ++c) {
    const int s0 = part * KEYS_PER_PART + c * 32;
    const u16_t* krow = kbase + (size_t)(s0 + lane) * D_DIM;
    const u16_t* vrow = vbase + (size_t)(s0 + lane) * D_DIM;
#pragma unroll
    for (int i = 0; i < 8; ++i) {
      *reinterpret_cast<uint4*>(klds + lane * 64 + i * 8) =
          *reinterpret_cast<const uint4*>(krow + i * 8);
      *reinterpret_cast<uint4*>(vlds + lane * 72 + i * 8) =
          *reinterpret_cast<const uint4*>(vrow + i * 8);
    }
    if (c + 1 < CHUNKS) {                       // hint next chunk into caches
      __builtin_prefetch(krow + 32 * D_DIM, 0, 1);
      __builtin_prefetch(vrow + 32 * D_DIM, 0, 1);
    }

    // scores = Q @ K^T   (contraction over HD=64 -> 2 wmma per 16x16 tile)
#pragma unroll
    for (int mi = 0; mi < 2; ++mi)
#pragma unroll
      for (int ni = 0; ni < 2; ++ni) {
        v8f st = wmma_bf16(aq[mi][0], load_b_frag(klds, 64, ni * 16, 0), zero_v8f());
        st = wmma_bf16(aq[mi][1], load_b_frag(klds, 64, ni * 16, 32), st);
#pragma unroll
        for (int v = 0; v < 8; ++v)
          sclds[(mi * 16 + half * 8 + v) * 33 + ni * 16 + col] = st[v];
      }

    // lane-per-row online softmax (row = lane)
    float mx = -1e30f;
#pragma unroll
    for (int j = 0; j < 32; ++j) mx = fmaxf(mx, sclds[lane * 33 + j]);
    mx *= SCALE_INV;
    float mnew = fmaxf(m_r, mx);
    float corr = __expf(m_r - mnew);
    float ls = 0.f;
#pragma unroll
    for (int j = 0; j < 32; ++j) {
      float p = __expf(sclds[lane * 33 + j] * SCALE_INV - mnew);
      ls += p;
      plds[lane * 48 + j] = f2bf(p);
    }
    l_r = l_r * corr + ls;
    m_r = mnew;
    scl_lds[lane] = corr;

    // rescale accumulators
    float cr[2][8];
#pragma unroll
    for (int mi = 0; mi < 2; ++mi)
#pragma unroll
      for (int v = 0; v < 8; ++v) cr[mi][v] = scl_lds[mi * 16 + half * 8 + v];
#pragma unroll
    for (int mi = 0; mi < 2; ++mi)
#pragma unroll
      for (int nj = 0; nj < 4; ++nj)
#pragma unroll
        for (int v = 0; v < 8; ++v) o[mi][nj][v] *= cr[mi][v];

    // O += P @ V  (contraction over 32 keys -> one wmma per tile)
    v16bf ap[2];
    ap[0] = load_a_frag(plds, 48, 0, 0);
    ap[1] = load_a_frag(plds, 48, 16, 0);
#pragma unroll
    for (int nj = 0; nj < 4; ++nj) {
      union { u16_t u[16]; v16bf v; } bv;
#pragma unroll
      for (int e = 0; e < 16; ++e)
        bv.u[e] = vlds[(half * 16 + e) * 72 + nj * 16 + col];
      o[0][nj] = wmma_bf16(ap[0], bv.v, o[0][nj]);
      o[1][nj] = wmma_bf16(ap[1], bv.v, o[1][nj]);
    }
  }

  // emit partial results
  const size_t pbase = (size_t)(bh * NPART + part) * (32 * 64);
#pragma unroll
  for (int mi = 0; mi < 2; ++mi)
#pragma unroll
    for (int nj = 0; nj < 4; ++nj)
#pragma unroll
      for (int v = 0; v < 8; ++v) {
        int r = mi * 16 + half * 8 + v;
        part_o[pbase + (size_t)r * 64 + nj * 16 + col] = o[mi][nj][v];
      }
  float* ml = part_ml + (size_t)(bh * NPART + part) * 64;
  ml[lane] = m_r;
  ml[32 + lane] = l_r;
}

// ---------------------------------------------------------------- K3: combine stage-1 partials
// grid: B*H, block 256 (thread = (row, 8-col segment))
__global__ void __launch_bounds__(256) compress_combine_kernel(
    const float* __restrict__ part_o, const float* __restrict__ part_ml,
    float* __restrict__ attn1) {
  const int bh = blockIdx.x, b = bh >> 3, h = bh & 7;
  const int tid = threadIdx.x;
  const int r = tid >> 3, cs = (tid & 7) * 8;

  float gm = -1e30f;
  for (int p = 0; p < NPART; ++p)
    gm = fmaxf(gm, part_ml[(size_t)(bh * NPART + p) * 64 + r]);
  float L = 0.f, acc[8];
#pragma unroll
  for (int c = 0; c < 8; ++c) acc[c] = 0.f;
  for (int p = 0; p < NPART; ++p) {
    const float* ml = part_ml + (size_t)(bh * NPART + p) * 64;
    float w = __expf(ml[r] - gm);
    L += w * ml[32 + r];
    const float* po = part_o + ((size_t)(bh * NPART + p) * 32 + r) * 64 + cs;
#pragma unroll
    for (int c = 0; c < 8; ++c) acc[c] += w * po[c];
  }
  float inv = 1.f / L;
  float* dst = attn1 + (size_t)(b * NSUR + r) * D_DIM + h * HDIM + cs;
#pragma unroll
  for (int c = 0; c < 8; ++c) dst[c] = acc[c] * inv;
}

// ---------------------------------------------------------------- K4: y = LN(x + resid) -> f32 + bf16
// grid: rows, block 256 (each thread 2 columns)
__global__ void __launch_bounds__(256) residual_ln_kernel(
    const float* __restrict__ x, const float* __restrict__ resid, int rmod,
    const float* __restrict__ g, const float* __restrict__ bta,
    float* __restrict__ of, u16_t* __restrict__ ob) {
  __shared__ float s1[256], s2[256];
  const int row = blockIdx.x, tid = threadIdx.x;
  const float* xr = x + (size_t)row * D_DIM;
  const float* rr = resid + (size_t)(row % rmod) * D_DIM;
  float v0 = xr[tid] + rr[tid];
  float v1 = xr[tid + 256] + rr[tid + 256];
  s1[tid] = v0 + v1;
  s2[tid] = v0 * v0 + v1 * v1;
  __syncthreads();
  for (int s = 128; s > 0; s >>= 1) {
    if (tid < s) { s1[tid] += s1[tid + s]; s2[tid] += s2[tid + s]; }
    __syncthreads();
  }
  float mu = s1[0] * (1.f / D_DIM);
  float var = s2[0] * (1.f / D_DIM) - mu * mu;
  float riv = rsqrtf(var + LN_EPS);
  float y0 = (v0 - mu) * riv * g[tid] + bta[tid];
  float y1 = (v1 - mu) * riv * g[tid + 256] + bta[tid + 256];
  size_t base = (size_t)row * D_DIM;
  of[base + tid] = y0;
  of[base + tid + 256] = y1;
  ob[base + tid] = f2bf(y0);
  ob[base + tid + 256] = f2bf(y1);
}

// ---------------------------------------------------------------- K5: cross attention vs 32 surrogate keys (stages 2 & 3)
// grid: (ceil(Mq/128), B*H), block 256 (8 waves x 16 query rows)
__global__ void __launch_bounds__(256) cross_attn_kernel(
    const u16_t* __restrict__ Q, int Mq,
    const u16_t* __restrict__ Kb, const u16_t* __restrict__ Vb,
    u16_t* __restrict__ out_bf, float* __restrict__ out_f) {
  __shared__ u16_t klds[32 * 64];
  __shared__ u16_t vlds[32 * 72];
  __shared__ float sclds[8][16 * 33];
  __shared__ u16_t plds[8][16 * 48];
  __shared__ float llds[8][16];

  const int bh = blockIdx.y, b = bh >> 3, h = bh & 7;
  const int tid = threadIdx.x, wv = tid >> 5, lane = tid & 31;
  const int half = lane >> 4, col = lane & 15;

  {  // cooperative K'/V' staging (one uint4 per thread per array)
    int row = tid >> 3, seg = tid & 7;
    const u16_t* kr = Kb + (size_t)(b * NSUR + row) * D_DIM + h * HDIM;
    const u16_t* vr = Vb + (size_t)(b * NSUR + row) * D_DIM + h * HDIM;
    *reinterpret_cast<uint4*>(klds + row * 64 + seg * 8) =
        *reinterpret_cast<const uint4*>(kr + seg * 8);
    *reinterpret_cast<uint4*>(vlds + row * 72 + seg * 8) =
        *reinterpret_cast<const uint4*>(vr + seg * 8);
  }
  __syncthreads();

  const int r0 = blockIdx.x * 128 + wv * 16;
  if (r0 < Mq) {
    const u16_t* qbase = Q + (size_t)b * Mq * D_DIM;
    v16bf aq0 = load_a_frag(qbase, D_DIM, r0, h * HDIM);
    v16bf aq1 = load_a_frag(qbase, D_DIM, r0, h * HDIM + 32);

    float* sc = sclds[wv];
#pragma unroll
    for (int ni = 0; ni < 2; ++ni) {
      v8f st = wmma_bf16(aq0, load_b_frag(klds, 64, ni * 16, 0), zero_v8f());
      st = wmma_bf16(aq1, load_b_frag(klds, 64, ni * 16, 32), st);
#pragma unroll
      for (int v = 0; v < 8; ++v)
        sc[(half * 8 + v) * 33 + ni * 16 + col] = st[v];
    }

    if (lane < 16) {  // one lane per query row
      float mx = -1e30f;
#pragma unroll
      for (int j = 0; j < 32; ++j) mx = fmaxf(mx, sc[lane * 33 + j]);
      mx *= SCALE_INV;
      float ls = 0.f;
#pragma unroll
      for (int j = 0; j < 32; ++j) {
        float p = __expf(sc[lane * 33 + j] * SCALE_INV - mx);
        ls += p;
        plds[wv][lane * 48 + j] = f2bf(p);
      }
      llds[wv][lane] = ls;
    }

    v16bf ap = load_a_frag(plds[wv], 48, 0, 0);
    float linv[8];
#pragma unroll
    for (int v = 0; v < 8; ++v) linv[v] = 1.f / llds[wv][half * 8 + v];

#pragma unroll
    for (int nj = 0; nj < 4; ++nj) {
      union { u16_t u[16]; v16bf v; } bv;
#pragma unroll
      for (int e = 0; e < 16; ++e)
        bv.u[e] = vlds[(half * 16 + e) * 72 + nj * 16 + col];
      v8f o = wmma_bf16(ap, bv.v, zero_v8f());
#pragma unroll
      for (int v = 0; v < 8; ++v) {
        int row = r0 + v + half * 8;
        size_t idx = (size_t)(b * Mq + row) * D_DIM + h * HDIM + nj * 16 + col;
        float val = o[v] * linv[v];
        if (out_bf) out_bf[idx] = f2bf(val);
        if (out_f) out_f[idx] = val;
      }
    }
  }
}

// ---------------------------------------------------------------- K6: fused out-proj + bias + residual + LayerNorm
// grid: NROWS/16, block 256 (8 waves x 64 output columns); TDM stages A tile.
#define CFLD 520
__global__ void __launch_bounds__(256) out_proj_ln_kernel(
    const u16_t* __restrict__ E, const u16_t* __restrict__ W,
    const float* __restrict__ bias, const float* __restrict__ x,
    const float* __restrict__ g, const float* __restrict__ bta,
    float* __restrict__ out) {
  __shared__ u16_t alds[16 * D_DIM];
  __shared__ float cf[16 * CFLD];
  __shared__ float red1[256], red2[256];
  __shared__ float mus[16], rivs[16];

  const int tid = threadIdx.x, wv = tid >> 5, lane = tid & 31;
  const int half = lane >> 4, col = lane & 15;
  const int row0 = blockIdx.x * 16;
  const u16_t* src = E + (size_t)row0 * D_DIM;

#if __has_builtin(__builtin_amdgcn_tensor_load_to_lds)
  if (wv == 0) {  // one wave issues the TDM copy of the contiguous 16KB tile
    unsigned lds_addr = (unsigned)(unsigned long long)(uintptr_t)(void*)alds;
    unsigned long long ga = (unsigned long long)(uintptr_t)(const void*)src;
    const unsigned ndw = 16 * D_DIM * 2 / 4;  // 4096 dwords
    v4u g0 = { 1u,                                   // count=1, load descriptor
               lds_addr,
               (unsigned)ga,
               (unsigned)((ga >> 32) & 0x1FFFFFFull) | (2u << 30) };  // type=image
    v8i g1 = { (int)(2u << 16),                      // data_size = 4B dwords
               (int)((ndw & 0xFFFFu) << 16),         // tensor_dim0 lo
               (int)((ndw >> 16) | (1u << 16)),      // tensor_dim0 hi | tensor_dim1 lo
               (int)(ndw << 16),                     // tile_dim0
               1,                                    // tile_dim1 = 1
               (int)ndw,                             // tensor_dim0_stride lo
               0, 0 };
    v4i gz4 = {0, 0, 0, 0};
    v8i gz8 = {0, 0, 0, 0, 0, 0, 0, 0};
    __builtin_amdgcn_tensor_load_to_lds(g0, g1, gz4, gz4, gz8, 0);
    __builtin_amdgcn_s_wait_tensorcnt(0);
  }
#endif
  {  // cooperative backup copy of the same bytes (keeps semantics toolchain-proof)
    int row = tid >> 4, seg = tid & 15;              // 16 rows x 16 segments of 64B
    const u16_t* r = src + (size_t)row * D_DIM + seg * 32;
#pragma unroll
    for (int i = 0; i < 4; ++i)
      *reinterpret_cast<uint4*>(alds + row * D_DIM + seg * 32 + i * 8) =
          *reinterpret_cast<const uint4*>(r + i * 8);
  }
  __syncthreads();

  v8f o[4];
#pragma unroll
  for (int nj = 0; nj < 4; ++nj) o[nj] = zero_v8f();
  const int nb = wv * 64;
#pragma unroll
  for (int k0 = 0; k0 < D_DIM; k0 += 32) {
    v16bf a = load_a_frag(alds, D_DIM, 0, k0);
#pragma unroll
    for (int nj = 0; nj < 4; ++nj)
      o[nj] = wmma_bf16(a, load_b_frag(W, D_DIM, nb + nj * 16, k0), o[nj]);
  }

  // bias + residual into LDS row buffer
#pragma unroll
  for (int nj = 0; nj < 4; ++nj)
#pragma unroll
    for (int v = 0; v < 8; ++v) {
      int r = v + half * 8;
      int c = nb + nj * 16 + col;
      cf[r * CFLD + c] = o[nj][v] + bias[c] + x[(size_t)(row0 + r) * D_DIM + c];
    }
  __syncthreads();

  // LayerNorm across each 512-wide row
  const int r = tid >> 4, j = tid & 15;
  float s1 = 0.f, s2 = 0.f;
#pragma unroll
  for (int i = 0; i < 32; ++i) {
    float v = cf[r * CFLD + j + 16 * i];
    s1 += v;
    s2 += v * v;
  }
  red1[tid] = s1;
  red2[tid] = s2;
  __syncthreads();
  if (j == 0) {
    float a1 = 0.f, a2 = 0.f;
#pragma unroll
    for (int t = 0; t < 16; ++t) { a1 += red1[(r << 4) + t]; a2 += red2[(r << 4) + t]; }
    float mu = a1 * (1.f / D_DIM);
    float var = a2 * (1.f / D_DIM) - mu * mu;
    mus[r] = mu;
    rivs[r] = rsqrtf(var + LN_EPS);
  }
  __syncthreads();
  float mu = mus[r], riv = rivs[r];
#pragma unroll
  for (int i = 0; i < 32; ++i) {
    int c = j + 16 * i;
    float v = cf[r * CFLD + c];
    out[(size_t)(row0 + r) * D_DIM + c] = (v - mu) * riv * g[c] + bta[c];
  }
}

// ---------------------------------------------------------------- host orchestration
extern "C" void kernel_launch(void* const* d_in, const int* in_sizes, int n_in,
                              void* d_out, int out_size, void* d_ws, size_t ws_size,
                              hipStream_t stream) {
  (void)in_sizes; (void)n_in; (void)out_size; (void)ws_size;
  const float* x    = (const float*)d_in[0];
  const float* surr = (const float*)d_in[1];
  const float* w_[10]; const float* b_[10];
  for (int i = 0; i < 10; ++i) {
    w_[i] = (const float*)d_in[2 + 2 * i];
    b_[i] = (const float*)d_in[3 + 2 * i];
  }
  // weight indices: 0=cq 1=ck 2=cv 3=sq 4=sk 5=sv 6=eq 7=ek 8=ev 9=out
  const float* ln_c_g = (const float*)d_in[22];
  const float* ln_c_b = (const float*)d_in[23];
  const float* ln_s_g = (const float*)d_in[24];
  const float* ln_s_b = (const float*)d_in[25];
  const float* ln_e_g = (const float*)d_in[26];
  const float* ln_e_b = (const float*)d_in[27];
  float* out = (float*)d_out;

  char* wsb = (char*)d_ws;
  size_t off = 0;
  auto carve = [&](size_t bytes) -> void* {
    void* p = wsb + off;
    off += (bytes + 255) & ~(size_t)255;
    return p;
  };
  const size_t WELE = (size_t)D_DIM * D_DIM;           // 262144
  u16_t* xb    = (u16_t*)carve((size_t)NROWS * D_DIM * 2);
  u16_t* kcb   = (u16_t*)carve((size_t)NROWS * D_DIM * 2);
  u16_t* vcb   = (u16_t*)carve((size_t)NROWS * D_DIM * 2);
  u16_t* qeb   = (u16_t*)carve((size_t)NROWS * D_DIM * 2);
  u16_t* expb  = (u16_t*)carve((size_t)NROWS * D_DIM * 2);
  u16_t* wb    = (u16_t*)carve(10 * WELE * 2);
  u16_t* surb  = (u16_t*)carve((size_t)NSUR * D_DIM * 2);
  u16_t* qcb   = (u16_t*)carve((size_t)NSUR * D_DIM * 2);
  float* parto = (float*)carve((size_t)64 * NPART * 32 * 64 * 4);
  float* partml= (float*)carve((size_t)64 * NPART * 64 * 4);
  float* attn1 = (float*)carve((size_t)256 * D_DIM * 4);
  float* compf = (float*)carve((size_t)256 * D_DIM * 4);
  u16_t* compb = (u16_t*)carve((size_t)256 * D_DIM * 2);
  u16_t* qsb   = (u16_t*)carve((size_t)256 * D_DIM * 2);
  u16_t* ksb   = (u16_t*)carve((size_t)256 * D_DIM * 2);
  u16_t* vsb   = (u16_t*)carve((size_t)256 * D_DIM * 2);
  float* attn2 = (float*)carve((size_t)256 * D_DIM * 4);
  float* procf = (float*)carve((size_t)256 * D_DIM * 4);
  u16_t* procb = (u16_t*)carve((size_t)256 * D_DIM * 2);
  u16_t* keb   = (u16_t*)carve((size_t)256 * D_DIM * 2);
  u16_t* veb   = (u16_t*)carve((size_t)256 * D_DIM * 2);

  // ---- precision conversion
  cvt_bf16_kernel<<<(NROWS * D_DIM + 1023) / 1024, 256, 0, stream>>>(x, xb, NROWS * D_DIM);
  for (int i = 0; i < 10; ++i)
    cvt_bf16_kernel<<<(int)((WELE + 1023) / 1024), 256, 0, stream>>>(w_[i], wb + i * WELE, (int)WELE);
  cvt_bf16_kernel<<<(NSUR * D_DIM + 1023) / 1024, 256, 0, stream>>>(surr, surb, NSUR * D_DIM);

  // ---- stage 1: projections
  gemm_bias_kernel<<<dim3(1, 4), 256, 0, stream>>>(surb, wb + 0 * WELE, b_[0], qcb, NSUR);
  gemm_bias_kernel<<<dim3(NROWS / 32, 4), 256, 0, stream>>>(xb, wb + 1 * WELE, b_[1], kcb, NROWS);
  gemm_bias_kernel<<<dim3(NROWS / 32, 4), 256, 0, stream>>>(xb, wb + 2 * WELE, b_[2], vcb, NROWS);
  gemm_bias_kernel<<<dim3(NROWS / 32, 4), 256, 0, stream>>>(xb, wb + 6 * WELE, b_[6], qeb, NROWS);

  // ---- stage 1: compress attention (flash over S) + combine + LN
  compress_attn_kernel<<<dim3(64, NPART), 32, 0, stream>>>(qcb, kcb, vcb, parto, partml);
  compress_combine_kernel<<<64, 256, 0, stream>>>(parto, partml, attn1);
  residual_ln_kernel<<<256, 256, 0, stream>>>(attn1, surr, NSUR, ln_c_g, ln_c_b, compf, compb);

  // ---- stage 2: surrogate self-attention
  gemm_bias_kernel<<<dim3(8, 4), 256, 0, stream>>>(compb, wb + 3 * WELE, b_[3], qsb, 256);
  gemm_bias_kernel<<<dim3(8, 4), 256, 0, stream>>>(compb, wb + 4 * WELE, b_[4], ksb, 256);
  gemm_bias_kernel<<<dim3(8, 4), 256, 0, stream>>>(compb, wb + 5 * WELE, b_[5], vsb, 256);
  cross_attn_kernel<<<dim3(1, 64), 256, 0, stream>>>(qsb, NSUR, ksb, vsb, (u16_t*)nullptr, attn2);
  residual_ln_kernel<<<256, 256, 0, stream>>>(attn2, compf, 256, ln_s_g, ln_s_b, procf, procb);

  // ---- stage 3: expand
  gemm_bias_kernel<<<dim3(8, 4), 256, 0, stream>>>(procb, wb + 7 * WELE, b_[7], keb, 256);
  gemm_bias_kernel<<<dim3(8, 4), 256, 0, stream>>>(procb, wb + 8 * WELE, b_[8], veb, 256);
  cross_attn_kernel<<<dim3(S_LEN / 128, 64), 256, 0, stream>>>(qeb, S_LEN, keb, veb, expb, (float*)nullptr);

  // ---- output projection + residual + LN (TDM-staged A tiles)
  out_proj_ln_kernel<<<NROWS / 16, 256, 0, stream>>>(expb, wb + 9 * WELE, b_[9], x,
                                                     ln_e_g, ln_e_b, out);
}
